// UnimodalBranch_31353261261529
// MI455X (gfx1250) — matrix operand order repaired and checked by reading, not compiled, not run on hardware
//
#include <hip/hip_runtime.h>
#include <hip/hip_bf16.h>
#include <cstddef>
#include <cstdint>

// ---------------------------------------------------------------------------
// Problem constants (from reference)
// ---------------------------------------------------------------------------
#define BB    8
#define CIN   64
#define HH    256
#define WW    256
#define COUT  128
#define H2    128
#define W2    128
#define KDIM  (CIN * 9)            // 576  (k = ci*9 + kh*3 + kw)
#define NPIX  (BB * H2 * W2)       // 131072 output pixels
#define NPTS  50000                // N

// Dynamic-LDS layout for the conv kernel:
//   [0, 5120)            : A tile, 64 x 40 bf16 (padded rows, 16B aligned)
//   [8192, 8192+147456)  : full bf16 weight panel [co][k], loaded by TDM
#define LDS_B_OFF   8192
#define LDS_TOTAL   (LDS_B_OFF + COUT * KDIM * 2)   // 155648 B (< 320 KB WGP LDS)

typedef __attribute__((ext_vector_type(16))) __bf16   v16bf;
typedef __attribute__((ext_vector_type(8)))  float    v8f;
typedef __attribute__((ext_vector_type(4)))  unsigned v4u;
typedef __attribute__((ext_vector_type(8)))  int      v8i;
typedef __attribute__((ext_vector_type(4)))  int      v4i;

union Frag16 {                      // 8 VGPRs = 16 bf16 per lane
    unsigned u[8];
    uint4    q[2];
    v16bf    v;
};

// ---------------------------------------------------------------------------
// Kernel 1: pack conv weights f32 (COUT,CIN,3,3) -> bf16 [co][k] (k inner
// linear order of w already equals ci*9+kh*3+kw, so this is a flat convert).
// ---------------------------------------------------------------------------
__global__ __launch_bounds__(256)
void pack_w_kernel(const float* __restrict__ w, __bf16* __restrict__ wp)
{
    int i = blockIdx.x * 256 + threadIdx.x;
    if (i < COUT * KDIM)
        wp[i] = (__bf16)w[i];
}

// ---------------------------------------------------------------------------
// Kernel 2: implicit-GEMM conv (stride 2, pad 1) + bias + ReLU via bf16 WMMA.
//   GEMM: [NPIX x KDIM] x [KDIM x COUT], f32 accumulate.
//   Block tile: 64 pixels x 128 channels; 8 waves; wave = 16(M) x 64(N).
//   B operand: the ENTIRE 144KB bf16 weight panel is DMA'd into LDS once by
//   the Tensor Data Mover (tensor_load_to_lds, TENSORcnt), so the K-loop only
//   stages the im2col A tile.
// ---------------------------------------------------------------------------
__global__ __launch_bounds__(256)
void conv_wmma_kernel(const float* __restrict__ x,
                      const __bf16* __restrict__ wp,
                      const float* __restrict__ bias,
                      float* __restrict__ y)
{
    extern __shared__ __align__(16) char smem[];
    __bf16* As = (__bf16*)smem;                 // [64][40]
    __bf16* Bw = (__bf16*)(smem + LDS_B_OFF);   // [COUT][KDIM]

    const int tid   = threadIdx.x;
    const int lane  = tid & 31;
    const int wid   = tid >> 5;          // 0..7
    const int half  = lane >> 4;         // 0/1: half-wave (K split in frags)
    const int l16   = lane & 15;
    const int m_off = (wid >> 1) << 4;   // 0,16,32,48
    const int n_off = (wid & 1) << 6;    // 0,64
    const int pix_base = blockIdx.x << 6;

    // ---- TDM prologue: DMA full weight panel global -> LDS (wave 0 only;
    // tensor ops are wave-level and ignore EXEC) ----
    if (wid == 0) {
        const unsigned long long ga = (unsigned long long)(uintptr_t)wp;
        // D# group0: count=1 | lds_addr | global_addr(57b) | type=2
        v4u g0 = { 1u,
                   (unsigned)LDS_B_OFF,
                   (unsigned)(ga & 0xffffffffu),
                   (unsigned)(((ga >> 32) & 0x01ffffffu) | 0x80000000u) };
        // D# group1: data_size=1 (2B elems); tensor_dim0=576, tensor_dim1=128;
        // tile_dim0=576, tile_dim1=128; tensor_dim0_stride=576; no pad/iterate.
        v8i g1 = { (int)0x00010000,
                   (int)(576u << 16),      // tensor_dim0[15:0] @ bits 63:48
                   (int)(128u << 16),      // tensor_dim1[15:0] @ bits 95:80
                   (int)(576u << 16),      // tile_dim0        @ bits 127:112
                   (int)128,               // tile_dim1        @ bits 143:128
                   (int)576,               // tensor_dim0_stride[31:0]
                   0, 0 };
        v4i gz4 = { 0, 0, 0, 0 };          // groups 2/3 unused (2-D tensor)
        v8i gz8 = { 0, 0, 0, 0, 0, 0, 0, 0 };
        __builtin_amdgcn_tensor_load_to_lds(g0, g1, gz4, gz4, gz8, 0);
        __builtin_amdgcn_s_wait_tensorcnt(0);
    }
    __syncthreads();                       // weight panel visible to all waves

    v8f acc[4] = {};                       // 4 x 16x16 f32 accumulators

    // A-tile loader coordinates: lane covers one k-column, 8 pixel rows.
    const int kk   = tid & 31;
    const int prow = tid >> 5;

    for (int k0 = 0; k0 < KDIM; k0 += 32) {
        __syncthreads();   // previous iteration's WMMA reads of As done

        // ---- stage A tile: im2col gather, f32 -> bf16, pad-zeroed ----
        {
            const int k  = k0 + kk;
            const int ci = k / 9;
            const int r9 = k - ci * 9;
            const int kh = r9 / 3;
            const int kw = r9 - kh * 3;
            #pragma unroll
            for (int pp = 0; pp < 8; ++pp) {
                const int pl = prow * 8 + pp;
                const int p  = pix_base + pl;
                const int b  = p >> 14;
                const int oh = (p >> 7) & 127;
                const int ow = p & 127;
                const int ih = oh * 2 - 1 + kh;
                const int iw = ow * 2 - 1 + kw;
                float v = 0.0f;
                if ((unsigned)ih < (unsigned)HH && (unsigned)iw < (unsigned)WW)
                    v = x[(((size_t)b * CIN + ci) * HH + ih) * WW + iw];
                As[pl * 40 + kk] = (__bf16)v;
            }
        }
        __syncthreads();

        // ---- fragments + WMMA chain ----
        Frag16 afr;
        {
            // 16-bit A 16x32 layout: VGPR0..3 hold K = half*8 + 0..7 (pairs),
            // VGPR4..7 hold K = 16 + half*8 + 0..7.
            const __bf16* ap = As + (m_off + l16) * 40;
            afr.q[0] = *reinterpret_cast<const uint4*>(ap + half * 8);
            afr.q[1] = *reinterpret_cast<const uint4*>(ap + 16 + half * 8);
        }
        #pragma unroll
        for (int nt = 0; nt < 4; ++nt) {
            // B 32x16: same K order as A (half*8 + 0..7, then 16 + half*8 + 0..7).
            Frag16 bfr;
            const __bf16* bp = Bw + (size_t)(n_off + nt * 16 + l16) * KDIM + k0;
            bfr.q[0] = *reinterpret_cast<const uint4*>(bp + half * 8);
            bfr.q[1] = *reinterpret_cast<const uint4*>(bp + 16 + half * 8);
            acc[nt] = __builtin_amdgcn_wmma_f32_16x16x32_bf16(
                          false, afr.v, false, bfr.v,
                          (short)0, acc[nt], false, false);
        }
    }

    // ---- epilogue: bias + ReLU, write y[pixel][cout] (reference `flat`) ----
    #pragma unroll
    for (int nt = 0; nt < 4; ++nt) {
        const int   co = n_off + nt * 16 + l16;
        const float bv = bias[co];
        #pragma unroll
        for (int r = 0; r < 8; ++r) {
            const int m = m_off + r + half * 8;   // C/D: VGPR r -> M = r (+8 hi half)
            float v = acc[nt][r] + bv;
            v = v > 0.0f ? v : 0.0f;
            y[(size_t)(pix_base + m) * COUT + co] = v;
        }
    }
}

// ---------------------------------------------------------------------------
// Kernel 3: fused gather + CSR segment-max (pixels->views->points) + add.
// One wave per point; lane holds 4 channels -> each row read is one fully
// coalesced 512B burst (global_load_b128 per lane).
// Empty view contributes 0; empty point yields 0 (matches isfinite masking).
// ---------------------------------------------------------------------------
__global__ __launch_bounds__(256)
void pool_fuse_kernel(const float* __restrict__ y,
                      const float* __restrict__ x3d,
                      const int* __restrict__ fm_idx,
                      const int* __restrict__ a_ptr,
                      const int* __restrict__ v_ptr,
                      float* __restrict__ out)
{
    const int wid  = threadIdx.x >> 5;
    const int lane = threadIdx.x & 31;
    const int n    = blockIdx.x * 8 + wid;   // 6250 * 8 == 50000 exactly
    const int c0   = lane << 2;

    const int v0 = v_ptr[n];
    const int v1 = v_ptr[n + 1];

    const float NEG = -__builtin_inff();
    float4 pmax = make_float4(NEG, NEG, NEG, NEG);

    for (int v = v0; v < v1; ++v) {
        const int a0 = a_ptr[v];
        const int a1 = a_ptr[v + 1];
        if (a0 < a1) {
            float4 vm = make_float4(NEG, NEG, NEG, NEG);
            for (int m = a0; m < a1; ++m) {
                const int idx = fm_idx[m];
                if (m + 1 < a1)
                    __builtin_prefetch(y + (size_t)fm_idx[m + 1] * COUT + c0, 0, 1);
                const float4 g =
                    *reinterpret_cast<const float4*>(y + (size_t)idx * COUT + c0);
                vm.x = fmaxf(vm.x, g.x);
                vm.y = fmaxf(vm.y, g.y);
                vm.z = fmaxf(vm.z, g.z);
                vm.w = fmaxf(vm.w, g.w);
            }
            pmax.x = fmaxf(pmax.x, vm.x);
            pmax.y = fmaxf(pmax.y, vm.y);
            pmax.z = fmaxf(pmax.z, vm.z);
            pmax.w = fmaxf(pmax.w, vm.w);
        } else {
            // empty view -> xv = 0 feeds the point max
            pmax.x = fmaxf(pmax.x, 0.0f);
            pmax.y = fmaxf(pmax.y, 0.0f);
            pmax.z = fmaxf(pmax.z, 0.0f);
            pmax.w = fmaxf(pmax.w, 0.0f);
        }
    }

    float4 xp;
    if (v0 == v1) {
        xp = make_float4(0.0f, 0.0f, 0.0f, 0.0f);   // empty point
    } else {
        xp = pmax;                                   // finite (ReLU outputs >= 0)
    }

    const size_t o = (size_t)n * COUT + c0;
    const float4 base = *reinterpret_cast<const float4*>(x3d + o);
    float4 res = make_float4(base.x + xp.x, base.y + xp.y,
                             base.z + xp.z, base.w + xp.w);
    *reinterpret_cast<float4*>(out + o) = res;
}

// ---------------------------------------------------------------------------
// Launch: inputs are (x, w, b, x_3d, fm_idx, a_ptr, v_ptr).
// Workspace layout: [0, 144KB) packed bf16 weights, [1MB, 1MB+64MB) conv
// output y in flat [pixel][cout] layout (needs ws_size >= ~69 MB).
// ---------------------------------------------------------------------------
extern "C" void kernel_launch(void* const* d_in, const int* in_sizes, int n_in,
                              void* d_out, int out_size, void* d_ws, size_t ws_size,
                              hipStream_t stream)
{
    const float* x      = (const float*)d_in[0];
    const float* w      = (const float*)d_in[1];
    const float* b      = (const float*)d_in[2];
    const float* x3d    = (const float*)d_in[3];
    const int*   fm_idx = (const int*)d_in[4];
    const int*   a_ptr  = (const int*)d_in[5];
    const int*   v_ptr  = (const int*)d_in[6];
    float*       out    = (float*)d_out;

    char*   ws = (char*)d_ws;
    __bf16* wp = (__bf16*)ws;                       // COUT*KDIM bf16 = 144 KB
    float*  y  = (float*)(ws + (size_t)(1 << 20));  // NPIX*COUT f32 = 64 MB

    {   // weight pack: 73728 elements
        const int total  = COUT * KDIM;
        const int blocks = (total + 255) / 256;
        pack_w_kernel<<<blocks, 256, 0, stream>>>(w, wp);
    }
    {   // conv: 131072 pixels / 64 per block; dynamic LDS = A tile + B panel
        conv_wmma_kernel<<<NPIX / 64, 256, LDS_TOTAL, stream>>>(x, wp, b, y);
    }
    {   // fused pooling: 8 points (waves) per block
        pool_fuse_kernel<<<NPTS / 8, 256, 0, stream>>>(y, x3d, fm_idx, a_ptr,
                                                       v_ptr, out);
    }
}
